// MolecularGNN_5059471475394
// MI455X (gfx1250) — matrix operand order repaired and checked by reading, not compiled, not run on hardware
//
#include <hip/hip_runtime.h>

typedef __attribute__((ext_vector_type(16))) _Float16 v16h;
typedef __attribute__((ext_vector_type(8)))  _Float16 v8h;
typedef __attribute__((ext_vector_type(4)))  _Float16 v4h;
typedef __attribute__((ext_vector_type(8)))  float    v8f;

#define HID       512
#define HEADS     8
#define HEAD_DIM  64
#define NNODES    50000
#define NEDGES    150000
#define ETOT      (NEDGES + NNODES)   /* edges + self loops */
#define NGRAPHS   2000
#define NLAYERS   6
#define NTARGETS  13
#define BN_EPS    1e-5f
#define NEG_SLOPE 0.2f

static inline int cdiv(int a, int b) { return (a + b - 1) / b; }

// ---------------------------------------------------------------- fill
__global__ void fill_kernel(float* p, float v, long n) {
  long i = (long)blockIdx.x * blockDim.x + threadIdx.x;
  long s = (long)gridDim.x * blockDim.x;
  for (; i < n; i += s) p[i] = v;
}

// ------------------------------------------------- WMMA GEMM (fp32 io, f16 MACs)
// C[M,N] = A[M,K] x B[K,N] (+bias[col]) (optional relu)
// block = 256 threads = 8 waves; block tile 128x128; wave tile 32x64 (2x4 WMMA)
// double-buffered LDS staging, fp32 -> f16 conversion during stage.
__global__ __launch_bounds__(256) void wmma_gemm_kernel(
    const float* __restrict__ A, const float* __restrict__ B,
    const float* __restrict__ bias, float* __restrict__ C,
    int M, int N, int K, int relu)
{
  __shared__ alignas(16) _Float16 As[2][128][40];  // row stride 80B: aligned frag reads
  __shared__ alignas(16) _Float16 Bt[2][128][40];  // B transposed (N-major)

  const int tid  = threadIdx.x;
  const int lane = tid & 31;
  const int wid  = tid >> 5;
  const int wm   = (wid >> 1) * 32;   // wave row offset in block tile (4 waves)
  const int wn   = (wid & 1)  * 64;   // wave col offset in block tile (2 waves)
  const int hlf  = lane >> 4;         // 0: lanes 0-15, 1: lanes 16-31
  const int l16  = lane & 15;
  const int mBase = blockIdx.x * 128;
  const int nBase = blockIdx.y * 128;

  const int kTiles = (K + 31) >> 5;

  // stage one 32-deep K slice (A: 128x32, B: 32x128 transposed) into buffer `buf`
  auto stage = [&](int kt, int buf) {
    const int k0 = kt << 5;
    // ---- A tile: thread -> (row = i>>3, 4-wide k chunk = (i&7)*4)
    if (mBase + 128 <= M && k0 + 32 <= K) {           // interior: vector fast path
      #pragma unroll
      for (int j = 0; j < 4; ++j) {
        int i = tid + (j << 8);
        int row = i >> 3, kg = (i & 7) << 2;
        const float4 f = *(const float4*)&A[(size_t)(mBase + row) * K + (k0 + kg)];
        v4h hv; hv.x = (_Float16)f.x; hv.y = (_Float16)f.y;
                hv.z = (_Float16)f.z; hv.w = (_Float16)f.w;
        *(v4h*)&As[buf][row][kg] = hv;
      }
    } else {                                           // edge: clamped branchless
      #pragma unroll
      for (int j = 0; j < 4; ++j) {
        int i = tid + (j << 8);
        int row = i >> 3, kg = (i & 7) << 2;
        int gr = mBase + row;
        float mr = (gr < M) ? 1.f : 0.f;
        int grc = gr < M ? gr : M - 1;
        v4h hv;
        #pragma unroll
        for (int q = 0; q < 4; ++q) {
          int gk = k0 + kg + q;
          float mk = (gk < K) ? mr : 0.f;
          int gkc = gk < K ? gk : K - 1;
          hv[q] = (_Float16)(A[(size_t)grc * K + gkc] * mk);
        }
        *(v4h*)&As[buf][row][kg] = hv;
      }
    }
    // ---- B tile (transposed): thread -> (col = i&127, 4-wide k chunk = (i>>7)*4)
    if (nBase + 128 <= N && k0 + 32 <= K) {
      #pragma unroll
      for (int j = 0; j < 4; ++j) {
        int i = tid + (j << 8);
        int col = i & 127, kg = (i >> 7) << 2;
        int gn = nBase + col;
        v4h hv;
        #pragma unroll
        for (int q = 0; q < 4; ++q)
          hv[q] = (_Float16)B[(size_t)(k0 + kg + q) * N + gn];
        *(v4h*)&Bt[buf][col][kg] = hv;
      }
    } else {
      #pragma unroll
      for (int j = 0; j < 4; ++j) {
        int i = tid + (j << 8);
        int col = i & 127, kg = (i >> 7) << 2;
        int gn = nBase + col;
        float mn = (gn < N) ? 1.f : 0.f;
        int gnc = gn < N ? gn : N - 1;
        v4h hv;
        #pragma unroll
        for (int q = 0; q < 4; ++q) {
          int gk = k0 + kg + q;
          float mk = (gk < K) ? mn : 0.f;
          int gkc = gk < K ? gk : K - 1;
          hv[q] = (_Float16)(B[(size_t)gkc * N + gnc] * mk);
        }
        *(v4h*)&Bt[buf][col][kg] = hv;
      }
    }
  };

  v8f acc[2][4] = {};

  stage(0, 0);
  __syncthreads();

  for (int kt = 0; kt < kTiles; ++kt) {
    const int cur = kt & 1;
    if (kt + 1 < kTiles) {
      stage(kt + 1, cur ^ 1);                          // overlap next stage w/ math
      if (kt + 2 < kTiles) {                           // speculative L2 prefetch
        int gr = mBase + (tid >> 1);
        if (gr < M) __builtin_prefetch(&A[(size_t)gr * K + ((kt + 2) << 5)], 0, 1);
      }
    }

    // fragments per CDNA5 16-bit A 16x32 layout:
    // lanes 0-15 hold K {0..7, 16..23}; lanes 16-31 hold K {8..15, 24..31}
    union Frag { v16h v; v8h h[2]; };
    Frag a[2], b[4];
    #pragma unroll
    for (int i = 0; i < 2; ++i) {
      int r = wm + i * 16 + l16;
      a[i].h[0] = *(const v8h*)&As[cur][r][hlf * 8];
      a[i].h[1] = *(const v8h*)&As[cur][r][16 + hlf * 8];
    }
    // B 32x16: lane = column; lanes 0-15 K {0..7, 16..23}, lanes 16-31 K {8..15, 24..31}
    #pragma unroll
    for (int j = 0; j < 4; ++j) {
      int c = wn + j * 16 + l16;
      b[j].h[0] = *(const v8h*)&Bt[cur][c][hlf * 8];
      b[j].h[1] = *(const v8h*)&Bt[cur][c][16 + hlf * 8];
    }
    #pragma unroll
    for (int i = 0; i < 2; ++i)
      #pragma unroll
      for (int j = 0; j < 4; ++j)
        acc[i][j] = __builtin_amdgcn_wmma_f32_16x16x32_f16(
            false, a[i].v, false, b[j].v, (short)0, acc[i][j], false, false);

    __syncthreads();   // next-stage data visible; cur buffer free for reuse
  }

  // C/D layout: VGPR v, lanes 0-15 -> row v, lanes 16-31 -> row v+8; col = lane%16
  #pragma unroll
  for (int i = 0; i < 2; ++i) {
    #pragma unroll
    for (int j = 0; j < 4; ++j) {
      int col = nBase + wn + j * 16 + l16;
      if (col < N) {
        float bv = bias ? bias[col] : 0.f;
        #pragma unroll
        for (int v = 0; v < 8; ++v) {
          int row = mBase + wm + i * 16 + hlf * 8 + v;
          if (row < M) {
            float x = acc[i][j][v] + bv;
            if (relu) x = fmaxf(x, 0.f);
            C[(size_t)row * N + col] = x;
          }
        }
      }
    }
  }
}

// ------------------------------------------------- per-node attention logits
// one wave per node: lane covers 16 contiguous channels (head = lane/4)
__global__ __launch_bounds__(256) void alpha_kernel(
    const float* __restrict__ h, const float* __restrict__ a_src,
    const float* __restrict__ a_dst, float* __restrict__ out_s,
    float* __restrict__ out_d, int n)
{
  int wid  = threadIdx.x >> 5;
  int lane = threadIdx.x & 31;
  int node = blockIdx.x * 8 + wid;
  if (node >= n) return;
  const float* hp = h + (size_t)node * HID + lane * 16;
  float s = 0.f, d = 0.f;
  #pragma unroll
  for (int i = 0; i < 16; ++i) {
    float v = hp[i];
    s += v * a_src[lane * 16 + i];
    d += v * a_dst[lane * 16 + i];
  }
  s += __shfl_xor(s, 1, 32); s += __shfl_xor(s, 2, 32);
  d += __shfl_xor(d, 1, 32); d += __shfl_xor(d, 2, 32);
  if ((lane & 3) == 0) {
    out_s[node * HEADS + (lane >> 2)] = s;
    out_d[node * HEADS + (lane >> 2)] = d;
  }
}

// ------------------------------------------------- float atomic max via int trick
__device__ inline void atomic_max_f32(float* addr, float val) {
  if (val >= 0.f) atomicMax((int*)addr, __float_as_int(val));
  else            atomicMin((unsigned int*)addr, (unsigned int)__float_as_int(val));
}

__device__ inline void edge_ends(int e, const int* src, const int* dst, int& s, int& d) {
  if (e < NEDGES) { s = src[e]; d = dst[e]; }
  else            { s = d = e - NEDGES; }   // self loop
}

__global__ void edge_max_kernel(const int* __restrict__ src, const int* __restrict__ dst,
                                const float* __restrict__ as_, const float* __restrict__ ad_,
                                float* __restrict__ emax)
{
  int idx = blockIdx.x * blockDim.x + threadIdx.x;
  if (idx >= ETOT * HEADS) return;
  int e = idx >> 3, hh = idx & 7;
  int s, d; edge_ends(e, src, dst, s, d);
  float v = as_[s * HEADS + hh] + ad_[d * HEADS + hh];
  v = v > 0.f ? v : NEG_SLOPE * v;
  atomic_max_f32(&emax[d * HEADS + hh], v);
}

__global__ void edge_exp_kernel(const int* __restrict__ src, const int* __restrict__ dst,
                                const float* __restrict__ as_, const float* __restrict__ ad_,
                                const float* __restrict__ emax, float* __restrict__ exbuf,
                                float* __restrict__ denom)
{
  int idx = blockIdx.x * blockDim.x + threadIdx.x;
  if (idx >= ETOT * HEADS) return;
  int e = idx >> 3, hh = idx & 7;
  int s, d; edge_ends(e, src, dst, s, d);
  float v = as_[s * HEADS + hh] + ad_[d * HEADS + hh];
  v = v > 0.f ? v : NEG_SLOPE * v;
  float ex = __expf(v - emax[d * HEADS + hh]);
  exbuf[idx] = ex;
  atomicAdd(&denom[d * HEADS + hh], ex);
}

// one wave per edge: lane l handles channels l, l+32, ... (coalesced)
__global__ __launch_bounds__(256) void aggregate_kernel(
    const int* __restrict__ src, const int* __restrict__ dst,
    const float* __restrict__ h, const float* __restrict__ exbuf,
    const float* __restrict__ denom, float* __restrict__ agg)
{
  int gw = (blockIdx.x * blockDim.x + threadIdx.x) >> 5;
  if (gw >= ETOT) return;
  int lane = threadIdx.x & 31;
  int s, d; edge_ends(gw, src, dst, s, d);
  float alpha = 0.f;
  if (lane < HEADS) alpha = exbuf[(size_t)gw * HEADS + lane] / denom[d * HEADS + lane];
  const float* hs = h + (size_t)s * HID;
  float* ag = agg + (size_t)d * HID;
  #pragma unroll
  for (int j = 0; j < 16; ++j) {
    int c = lane + (j << 5);
    float al = __shfl(alpha, c >> 6, 32);
    atomicAdd(&ag[c], hs[c] * al);
  }
}

// ------------------------------------------------- batchnorm (gat_b cancels)
__global__ __launch_bounds__(512) void bn_stats_kernel(
    const float* __restrict__ x, float* __restrict__ sum, float* __restrict__ sq, int n)
{
  int c = threadIdx.x;
  float s = 0.f, q = 0.f;
  for (int r = blockIdx.x; r < n; r += gridDim.x) {
    float v = x[(size_t)r * HID + c];
    s += v; q += v * v;
  }
  atomicAdd(&sum[c], s);
  atomicAdd(&sq[c], q);
}

__global__ void bn_finalize_kernel(const float* sum, const float* sq,
                                   const float* gamma, const float* beta,
                                   float* scale, float* shift, int n)
{
  int c = threadIdx.x;
  float inv_n = 1.f / (float)n;
  float mu  = sum[c] * inv_n;
  float var = sq[c] * inv_n - mu * mu;
  float sc  = gamma[c] * rsqrtf(var + BN_EPS);
  scale[c] = sc;
  shift[c] = beta[c] - mu * sc;
}

__global__ void bn_apply_kernel(float* x, const float* __restrict__ scale,
                                const float* __restrict__ shift, long total)
{
  long i = (long)blockIdx.x * blockDim.x + threadIdx.x;
  long s = (long)gridDim.x * blockDim.x;
  for (; i < total; i += s) {
    int c = (int)(i & (HID - 1));
    x[i] = fmaxf(x[i] * scale[c] + shift[c], 0.f);
  }
}

// ------------------------------------------------- global mean pool
__global__ void pool_sum_kernel(const float* __restrict__ x, const int* __restrict__ batch,
                                float* __restrict__ pooled, long total)
{
  long i = (long)blockIdx.x * blockDim.x + threadIdx.x;
  long s = (long)gridDim.x * blockDim.x;
  for (; i < total; i += s) {
    int n = (int)(i >> 9), c = (int)(i & 511);
    atomicAdd(&pooled[(size_t)batch[n] * HID + c], x[i]);
  }
}

__global__ void pool_count_kernel(const int* __restrict__ batch, float* counts, int n) {
  int i = blockIdx.x * blockDim.x + threadIdx.x;
  if (i < n) atomicAdd(&counts[batch[i]], 1.f);
}

__global__ void pool_div_kernel(float* pooled, const float* __restrict__ counts, int total) {
  int i = blockIdx.x * blockDim.x + threadIdx.x;
  if (i < total) pooled[i] /= fmaxf(counts[i >> 9], 1.f);
}

// ----------------------------------------------------------------
extern "C" void kernel_launch(void* const* d_in, const int* in_sizes, int n_in,
                              void* d_out, int out_size, void* d_ws, size_t ws_size,
                              hipStream_t stream)
{
  const float* atom_features = (const float*)d_in[0];
  /* d_in[1] bond_features: unused by the reference model */
  const int*   edge_index = (const int*)d_in[2];
  const int*   batch      = (const int*)d_in[3];
  const float* atom_w  = (const float*)d_in[4];
  const float* atom_b  = (const float*)d_in[5];
  const float* gat_w   = (const float*)d_in[6];
  const float* att_src = (const float*)d_in[7];
  const float* att_dst = (const float*)d_in[8];
  /* d_in[9] gat_b: algebraically cancels inside batchnorm */
  const float* bn_gamma = (const float*)d_in[10];
  const float* bn_beta  = (const float*)d_in[11];
  const float* out_w1 = (const float*)d_in[12];
  const float* out_b1 = (const float*)d_in[13];
  const float* out_w2 = (const float*)d_in[14];
  const float* out_b2 = (const float*)d_in[15];

  float* p = (float*)d_ws;
  float* xbuf   = p; p += (size_t)NNODES * HID;     // layer input / aggregation target
  float* hbuf   = p; p += (size_t)NNODES * HID;     // W-projected features
  float* as_    = p; p += NNODES * HEADS;
  float* ad_    = p; p += NNODES * HEADS;
  float* emax   = p; p += NNODES * HEADS;
  float* denom  = p; p += NNODES * HEADS;
  float* exbuf  = p; p += (size_t)ETOT * HEADS;
  float* bnsum  = p; p += HID;
  float* bnsq   = p; p += HID;
  float* bnscale = p; p += HID;
  float* bnshift = p; p += HID;
  float* pooled = p; p += (size_t)NGRAPHS * HID;
  float* counts = p; p += NGRAPHS;
  float* hidden = p; p += (size_t)NGRAPHS * 256;
  (void)ws_size; (void)n_in; (void)in_sizes; (void)out_size;

  const int* e_src = edge_index;
  const int* e_dst = edge_index + NEDGES;

  // atom projection: [50000,100] @ [100,512] + atom_b
  wmma_gemm_kernel<<<dim3(cdiv(NNODES, 128), cdiv(HID, 128)), 256, 0, stream>>>(
      atom_features, atom_w, atom_b, xbuf, NNODES, HID, 100, 0);

  for (int L = 0; L < NLAYERS; ++L) {
    // h = x @ W   [50000,512]@[512,512]
    wmma_gemm_kernel<<<dim3(cdiv(NNODES, 128), cdiv(HID, 128)), 256, 0, stream>>>(
        xbuf, gat_w + (size_t)L * HID * HID, nullptr, hbuf, NNODES, HID, HID, 0);
    // per-node attention logits
    alpha_kernel<<<cdiv(NNODES, 8), 256, 0, stream>>>(
        hbuf, att_src + L * HID, att_dst + L * HID, as_, ad_, NNODES);
    // edge softmax + aggregation into (reused) xbuf
    fill_kernel<<<1024, 256, 0, stream>>>(emax, -3.0e38f, (long)NNODES * HEADS);
    fill_kernel<<<1024, 256, 0, stream>>>(denom, 0.f, (long)NNODES * HEADS);
    fill_kernel<<<2048, 256, 0, stream>>>(xbuf, 0.f, (long)NNODES * HID);
    edge_max_kernel<<<cdiv(ETOT * HEADS, 256), 256, 0, stream>>>(e_src, e_dst, as_, ad_, emax);
    edge_exp_kernel<<<cdiv(ETOT * HEADS, 256), 256, 0, stream>>>(e_src, e_dst, as_, ad_, emax, exbuf, denom);
    aggregate_kernel<<<cdiv(ETOT * 32, 256), 256, 0, stream>>>(e_src, e_dst, hbuf, exbuf, denom, xbuf);
    // batchnorm + relu, in place on xbuf
    fill_kernel<<<2, 256, 0, stream>>>(bnsum, 0.f, HID);
    fill_kernel<<<2, 256, 0, stream>>>(bnsq, 0.f, HID);
    bn_stats_kernel<<<256, 512, 0, stream>>>(xbuf, bnsum, bnsq, NNODES);
    bn_finalize_kernel<<<1, 512, 0, stream>>>(bnsum, bnsq, bn_gamma + L * HID,
                                              bn_beta + L * HID, bnscale, bnshift, NNODES);
    bn_apply_kernel<<<2048, 256, 0, stream>>>(xbuf, bnscale, bnshift, (long)NNODES * HID);
  }

  // global mean pool
  fill_kernel<<<512, 256, 0, stream>>>(pooled, 0.f, (long)NGRAPHS * HID);
  fill_kernel<<<8, 256, 0, stream>>>(counts, 0.f, NGRAPHS);
  pool_sum_kernel<<<2048, 256, 0, stream>>>(xbuf, batch, pooled, (long)NNODES * HID);
  pool_count_kernel<<<cdiv(NNODES, 256), 256, 0, stream>>>(batch, counts, NNODES);
  pool_div_kernel<<<cdiv(NGRAPHS * HID, 256), 256, 0, stream>>>(pooled, counts, NGRAPHS * HID);

  // head MLP
  wmma_gemm_kernel<<<dim3(cdiv(NGRAPHS, 128), cdiv(256, 128)), 256, 0, stream>>>(
      pooled, out_w1, out_b1, hidden, NGRAPHS, 256, HID, 1);
  wmma_gemm_kernel<<<dim3(cdiv(NGRAPHS, 128), 1), 256, 0, stream>>>(
      hidden, out_w2, out_b2, (float*)d_out, NGRAPHS, NTARGETS, 256, 0);
}
